// Qint_Linear_18786186953100
// MI455X (gfx1250) — compile-verified, matrix operand order checked by reading, not run on hardware
//
#include <hip/hip_runtime.h>

#define IN_F   4096
#define OUT_F  11008
#define TOKENS 4096

typedef __attribute__((ext_vector_type(16))) _Float16 v16h;
typedef __attribute__((ext_vector_type(8)))  _Float16 v8h;
typedef __attribute__((ext_vector_type(4)))  _Float16 v4h;
typedef __attribute__((ext_vector_type(8)))  float    v8f;
typedef __attribute__((ext_vector_type(4)))  float    v4f;
typedef __attribute__((ext_vector_type(4)))  int      v4i;

// ---------------------------------------------------------------------------
// CDNA5 async global->LDS copy (16B per lane), tracked by ASYNCcnt.
// ---------------------------------------------------------------------------
__device__ __forceinline__ void async_copy16(void* lds_ptr, const void* gptr) {
    // low 32 bits of a flat LDS pointer are the workgroup-relative LDS offset
    unsigned l = (unsigned)(uintptr_t)lds_ptr;
    asm volatile("global_load_async_to_lds_b128 %0, %1, off"
                 :: "v"(l), "v"(gptr) : "memory");
}
__device__ __forceinline__ void wait_async_all() {
    asm volatile("s_wait_asynccnt 0" ::: "memory");
}

// ---------------------------------------------------------------------------
// Prepass kernels: one-shot conversion into f16 workspace.
// ---------------------------------------------------------------------------
__global__ __launch_bounds__(256) void cvt_f32_to_f16(const float* __restrict__ in,
                                                      _Float16* __restrict__ out) {
    size_t i = ((size_t)blockIdx.x * 256 + threadIdx.x) * 8;
    v4f a = *(const v4f*)(in + i);
    v4f b = *(const v4f*)(in + i + 4);
    v8h h = { (_Float16)a[0], (_Float16)a[1], (_Float16)a[2], (_Float16)a[3],
              (_Float16)b[0], (_Float16)b[1], (_Float16)b[2], (_Float16)b[3] };
    *(v8h*)(out + i) = h;
}

__global__ __launch_bounds__(256) void cvt_i32_to_f16(const int* __restrict__ in,
                                                      _Float16* __restrict__ out) {
    size_t i = ((size_t)blockIdx.x * 256 + threadIdx.x) * 8;
    v4i a = *(const v4i*)(in + i);
    v4i b = *(const v4i*)(in + i + 4);
    v8h h = { (_Float16)a[0], (_Float16)a[1], (_Float16)a[2], (_Float16)a[3],
              (_Float16)b[0], (_Float16)b[1], (_Float16)b[2], (_Float16)b[3] };
    *(v8h*)(out + i) = h;
}

// ---------------------------------------------------------------------------
// Main GEMM: f16 operands from workspace, async-LDS double buffering.
// Block = 128(M) x 256(N), 8 waves, each wave computes 64x64.
// BK = 32 (one wmma k-step x ... 16 wmma per wave per slab).
// ---------------------------------------------------------------------------
constexpr int GBM = 128;
constexpr int GBN = 256;
constexpr int GBK = 32;
constexpr int GLD = GBK + 8;   // LDS row stride in halves (80B, 16B-aligned)

__global__ __launch_bounds__(256) void qlin_wmma_async(
    const _Float16* __restrict__ x16,  // [TOKENS, IN_F]
    const _Float16* __restrict__ w16,  // [OUT_F, IN_F]
    const float* __restrict__ scales,
    const float* __restrict__ bias,
    float* __restrict__ out)
{
    __shared__ __align__(16) _Float16 As[2 * GBM * GLD];   // 2 x 10 KB
    __shared__ __align__(16) _Float16 Bs[2 * GBN * GLD];   // 2 x 20 KB

    const int tid  = threadIdx.x;
    const int m0   = blockIdx.y * GBM;
    const int n0   = blockIdx.x * GBN;

    const int wave    = tid >> 5;
    const int lane    = tid & 31;
    const int waveM   = wave & 1;     // 2 waves along M -> 64 rows each
    const int waveN   = wave >> 1;    // 4 waves along N -> 64 cols each
    const int half_id = lane >> 4;
    const int r       = lane & 15;

    v8f acc[4][4];
    #pragma unroll
    for (int i = 0; i < 4; ++i)
        #pragma unroll
        for (int j = 0; j < 4; ++j)
            acc[i][j] = (v8f)0.0f;

    // ---- async stage one k-slab into buffer `buf` -------------------------
    // A: 128 rows x 4 16B-chunks = 512 chunks; B: 256 x 4 = 1024; 6/thread.
    auto issue_async = [&](int buf, int k0) {
        _Float16* Ab = As + buf * (GBM * GLD);
        _Float16* Bb = Bs + buf * (GBN * GLD);
        #pragma unroll
        for (int i = 0; i < 2; ++i) {
            const int c   = tid + i * 256;     // 0..511
            const int row = c >> 2;
            const int ch  = c & 3;
            async_copy16(Ab + row * GLD + ch * 8,
                         x16 + (size_t)(m0 + row) * IN_F + k0 + ch * 8);
        }
        #pragma unroll
        for (int i = 0; i < 4; ++i) {
            const int c   = tid + i * 256;     // 0..1023
            const int row = c >> 2;
            const int ch  = c & 3;
            async_copy16(Bb + row * GLD + ch * 8,
                         w16 + (size_t)(n0 + row) * IN_F + k0 + ch * 8);
        }
    };

    // ---- compute one staged slab: 16 wmma per wave ------------------------
    auto compute = [&](int buf) {
        const _Float16* Ab = As + buf * (GBM * GLD);
        const _Float16* Bb = Bs + buf * (GBN * GLD);
        v16h af[4];
        #pragma unroll
        for (int mi = 0; mi < 4; ++mi) {
            const _Float16* p = Ab + (waveM * 64 + mi * 16 + r) * GLD + half_id * 8;
            v8h lo = *(const v8h*)(p);
            v8h hi = *(const v8h*)(p + 16);
            af[mi] = __builtin_shufflevector(lo, hi,
                0,1,2,3,4,5,6,7,8,9,10,11,12,13,14,15);
        }
        #pragma unroll
        for (int ni = 0; ni < 4; ++ni) {
            const _Float16* p = Bb + (waveN * 64 + ni * 16 + r) * GLD + half_id * 8;
            v8h lo = *(const v8h*)(p);
            v8h hi = *(const v8h*)(p + 16);
            v16h bf = __builtin_shufflevector(lo, hi,
                0,1,2,3,4,5,6,7,8,9,10,11,12,13,14,15);
            #pragma unroll
            for (int mi = 0; mi < 4; ++mi) {
                acc[mi][ni] = __builtin_amdgcn_wmma_f32_16x16x32_f16(
                    false, af[mi], false, bf, (short)0, acc[mi][ni], false, false);
            }
        }
    };

    // ---- pipelined k loop: async DMA of slab t+1 overlaps wmma on slab t --
    issue_async(0, 0);
    wait_async_all();
    __syncthreads();

    int buf = 0;
    for (int k0 = GBK; k0 < IN_F; k0 += GBK) {
        issue_async(buf ^ 1, k0);  // DMA in flight during compute
        compute(buf);
        wait_async_all();          // this wave's DMA done
        __syncthreads();           // LDS visible / reads of old buf done
        buf ^= 1;
    }
    compute(buf);

    // ---- epilogue: per-column scale + bias --------------------------------
    #pragma unroll
    for (int ni = 0; ni < 4; ++ni) {
        const int col = n0 + waveN * 64 + ni * 16 + r;
        const float s = scales[col];
        const float b = bias[col];
        #pragma unroll
        for (int mi = 0; mi < 4; ++mi) {
            const int rowbase = m0 + waveM * 64 + mi * 16 + half_id * 8;
            #pragma unroll
            for (int v = 0; v < 8; ++v) {
                out[(size_t)(rowbase + v) * OUT_F + col] = acc[mi][ni][v] * s + b;
            }
        }
    }
}

// ---------------------------------------------------------------------------
// Fallback (ws too small): round-1 fused-conversion kernel, 128x128 blocks.
// ---------------------------------------------------------------------------
constexpr int FBM = 128, FBN = 128, FBK = 32, FLD = FBK + 8;

__global__ __launch_bounds__(256) void qlin_wmma_fused(
    const float* __restrict__ x, const int* __restrict__ w,
    const float* __restrict__ scales, const float* __restrict__ bias,
    float* __restrict__ out)
{
    __shared__ __align__(16) _Float16 As[FBM * FLD];
    __shared__ __align__(16) _Float16 Bs[FBN * FLD];

    const int tid = threadIdx.x;
    const int m0 = blockIdx.y * FBM;
    const int n0 = blockIdx.x * FBN;
    const int wave = tid >> 5, lane = tid & 31;
    const int waveM = wave & 3, waveN = wave >> 2;
    const int half_id = lane >> 4, r = lane & 15;

    v8f acc[2][4];
    #pragma unroll
    for (int i = 0; i < 2; ++i)
        #pragma unroll
        for (int j = 0; j < 4; ++j) acc[i][j] = (v8f)0.0f;

    v4f aReg[4]; v4i bReg[4];
    auto load_global = [&](int k0) {
        #pragma unroll
        for (int i = 0; i < 4; ++i) {
            const int f = tid + i * 256, row = f >> 3, c4 = f & 7;
            aReg[i] = *(const v4f*)(x + (size_t)(m0 + row) * IN_F + k0 + c4 * 4);
            bReg[i] = *(const v4i*)(w + (size_t)(n0 + row) * IN_F + k0 + c4 * 4);
        }
    };
    auto store_lds = [&]() {
        #pragma unroll
        for (int i = 0; i < 4; ++i) {
            const int f = tid + i * 256, row = f >> 3, c4 = f & 7;
            v4h ah = { (_Float16)aReg[i][0], (_Float16)aReg[i][1],
                       (_Float16)aReg[i][2], (_Float16)aReg[i][3] };
            v4h bh = { (_Float16)bReg[i][0], (_Float16)bReg[i][1],
                       (_Float16)bReg[i][2], (_Float16)bReg[i][3] };
            *(v4h*)(As + row * FLD + c4 * 4) = ah;
            *(v4h*)(Bs + row * FLD + c4 * 4) = bh;
        }
    };
    auto compute = [&]() {
        v16h af[2];
        #pragma unroll
        for (int mi = 0; mi < 2; ++mi) {
            const _Float16* p = As + (waveM * 32 + mi * 16 + r) * FLD + half_id * 8;
            v8h lo = *(const v8h*)(p); v8h hi = *(const v8h*)(p + 16);
            af[mi] = __builtin_shufflevector(lo, hi, 0,1,2,3,4,5,6,7,8,9,10,11,12,13,14,15);
        }
        #pragma unroll
        for (int ni = 0; ni < 4; ++ni) {
            const _Float16* p = Bs + (waveN * 64 + ni * 16 + r) * FLD + half_id * 8;
            v8h lo = *(const v8h*)(p); v8h hi = *(const v8h*)(p + 16);
            v16h bf = __builtin_shufflevector(lo, hi, 0,1,2,3,4,5,6,7,8,9,10,11,12,13,14,15);
            #pragma unroll
            for (int mi = 0; mi < 2; ++mi)
                acc[mi][ni] = __builtin_amdgcn_wmma_f32_16x16x32_f16(
                    false, af[mi], false, bf, (short)0, acc[mi][ni], false, false);
        }
    };

    load_global(0); store_lds(); __syncthreads();
    for (int k0 = FBK; k0 < IN_F; k0 += FBK) {
        load_global(k0);
        compute();
        __syncthreads();
        store_lds();
        __syncthreads();
    }
    compute();

    #pragma unroll
    for (int ni = 0; ni < 4; ++ni) {
        const int col = n0 + waveN * 64 + ni * 16 + r;
        const float s = scales[col], b = bias[col];
        #pragma unroll
        for (int mi = 0; mi < 2; ++mi) {
            const int rowbase = m0 + waveM * 32 + mi * 16 + half_id * 8;
            #pragma unroll
            for (int v = 0; v < 8; ++v)
                out[(size_t)(rowbase + v) * OUT_F + col] = acc[mi][ni][v] * s + b;
        }
    }
}

// ---------------------------------------------------------------------------
extern "C" void kernel_launch(void* const* d_in, const int* in_sizes, int n_in,
                              void* d_out, int out_size, void* d_ws, size_t ws_size,
                              hipStream_t stream) {
    const float* x      = (const float*)d_in[0];
    const int*   w      = (const int*)  d_in[1];
    const float* scales = (const float*)d_in[2];
    const float* bias   = (const float*)d_in[3];
    float*       out    = (float*)d_out;

    const size_t xN = (size_t)TOKENS * IN_F;          // 16,777,216
    const size_t wN = (size_t)OUT_F  * IN_F;          // 45,088,768
    const size_t needed = (xN + wN) * sizeof(_Float16);

    if (ws_size >= needed) {
        _Float16* x16 = (_Float16*)d_ws;
        _Float16* w16 = x16 + xN;
        cvt_f32_to_f16<<<(int)(xN / 2048), 256, 0, stream>>>(x, x16);
        cvt_i32_to_f16<<<(int)(wN / 2048), 256, 0, stream>>>(w, w16);
        dim3 grid(OUT_F / GBN, TOKENS / GBM);         // 43 x 32
        qlin_wmma_async<<<grid, 256, 0, stream>>>(x16, w16, scales, bias, out);
    } else {
        dim3 grid(OUT_F / FBN, TOKENS / FBM);         // 86 x 32
        qlin_wmma_fused<<<grid, 256, 0, stream>>>(x, w, scales, bias, out);
    }
}
